// CGConvBlock_15848429322413
// MI455X (gfx1250) — matrix operand order repaired and checked by reading, not compiled, not run on hardware
//
#include <hip/hip_runtime.h>

#define Nn 20000
#define Ee 640000
#define Cc 128
#define Dd 64
#define Hh 512
#define Ll 3
#define Gg 32
#define Zz 320           // 2C + D
#define EPSf 1e-5f

typedef __attribute__((ext_vector_type(16))) __bf16 v16bf;
typedef __attribute__((ext_vector_type(8)))  __bf16 v8bf;
typedef __attribute__((ext_vector_type(8)))  float  v8f;
typedef __attribute__((ext_vector_type(4)))  float  v4f;
typedef __attribute__((ext_vector_type(4)))  unsigned short v4us;

union ABfrag { v16bf v; v8bf h[2]; };

__device__ __forceinline__ unsigned short bfbits(float f) {
    unsigned u = __builtin_bit_cast(unsigned, f);
    u += 0x7FFFu + ((u >> 16) & 1u);
    return (unsigned short)(u >> 16);
}
__device__ __forceinline__ __bf16 f2bf(float f) {
    return __builtin_bit_cast(__bf16, bfbits(f));
}
__device__ __forceinline__ void store4bf(__bf16* p, v4f v) {
    v4us h;
    h[0] = bfbits(v[0]); h[1] = bfbits(v[1]);
    h[2] = bfbits(v[2]); h[3] = bfbits(v[3]);
    *(v4us*)p = h;
}

// async 16B global->LDS copy (CDNA5).  lds_off = LDS byte address (low 32 bits
// of generic pointer, per aperture rule); gaddr = 64-bit global address.
__device__ __forceinline__ void async_copy_b128(unsigned lds_off, const void* g) {
    asm volatile("global_load_async_to_lds_b128 %0, %1, off"
                 :: "v"(lds_off), "v"((unsigned long long)(size_t)g)
                 : "memory");
}
__device__ __forceinline__ void async_wait0() {
    asm volatile("s_wait_asynccnt 0x0" ::: "memory");
}

// ---------------------------------------------------------------------------
// Pack row-major W (outC x K, f32) into CDNA5 B-fragment layout (bf16):
// [nt][kc][lane][i] with n = nt*16 + (lane&15), k = kc*32 + (lane>>4)*8 + (i&7) + (i>>3)*16
// ---------------------------------------------------------------------------
__global__ void pack_b_kernel(const float* __restrict__ W, __bf16* __restrict__ out,
                              int outC, int K) {
    int tid = blockIdx.x * 256 + threadIdx.x;
    int total = outC * K;
    if (tid >= total) return;
    int i    = tid & 15;
    int lane = (tid >> 4) & 31;
    int kcN  = K >> 5;
    int kc   = (tid >> 9) % kcN;
    int nt   = tid / (kcN << 9);
    int n = nt * 16 + (lane & 15);
    int k = kc * 32 + (lane >> 4) * 8 + (i & 7) + (i >> 3) * 16;
    out[tid] = f2bf(W[(size_t)n * K + k]);
}

// ---------------------------------------------------------------------------
// Small utility kernels
// ---------------------------------------------------------------------------
__global__ void zero_kernel(float* p, int n) {
    int i = blockIdx.x * 256 + threadIdx.x;
    if (i < n) p[i] = 0.f;
}
__global__ void count_kernel(const int* __restrict__ nb, float* cnt) {
    int i = blockIdx.x * 256 + threadIdx.x;
    if (i < Nn) atomicAdd(&cnt[nb[i]], 1.f);
}
// x += agg, and produce bf16 shadow for the next GEMM's A staging
__global__ void addagg_kernel(float* xb, __bf16* __restrict__ xbf,
                              const float* __restrict__ agg, int n) {
    int i = blockIdx.x * 256 + threadIdx.x;
    if (i < n) {
        float v = xb[i] + agg[i];
        xb[i] = v;
        xbf[i] = f2bf(v);
    }
}
// f32 -> bf16 bulk convert (4 elems/thread)
__global__ void cvt_bf16_kernel(const float* __restrict__ in, __bf16* __restrict__ out, int n4) {
    int i = blockIdx.x * 256 + threadIdx.x;
    if (i < n4) store4bf(out + (size_t)i * 4, *(const v4f*)(in + (size_t)i * 4));
}

// ---------------------------------------------------------------------------
// Edge kernel: per 16-edge tile, z = [xbf[dst], xbf[src], eabf] staged into
// LDS entirely with GLOBAL_LOAD_ASYNC_TO_LDS_B128 (ASYNCcnt), then
// gate = sigmoid(z Wf^T + bf), core = softplus(z Ws^T + bs),
// atomic scatter of gate*core into agg[dst].
// One wave per tile, 4 waves/block. 40000 tiles.
// LDS row = 656 B (41 x 16B chunks): 16B-aligned, bank-conflict-free frags.
// ---------------------------------------------------------------------------
__global__ void edge_kernel(const __bf16* __restrict__ xbf, const __bf16* __restrict__ eabf,
                            const int* __restrict__ eidx,
                            const __bf16* __restrict__ packFS,
                            const float* __restrict__ bfv, const float* __restrict__ bsv,
                            float* __restrict__ agg, int layer) {
    const int wave = threadIdx.x >> 5;
    const int lane = threadIdx.x & 31;
    const int tile = blockIdx.x * 4 + wave;

    __shared__ __bf16 zbuf[4][16][328];   // 328 halves = 656 B per row
    __shared__ int sdst[4][16];
    __shared__ int ssrc[4][16];

    const int e0 = tile * 16;
    if (lane < 16)  sdst[wave][lane]      = eidx[Ee + e0 + lane];   // edge_index[1] = dst
    if (lane >= 16) ssrc[wave][lane - 16] = eidx[e0 + lane - 16];   // edge_index[0] = src

    // ---- async-stage z tile: 16 rows x 40 chunks of 16B = 640 chunks ----
    const unsigned ldsbase = (unsigned)(size_t)&zbuf[wave][0][0];
#pragma unroll
    for (int t = 0; t < 20; ++t) {
        int c = t * 32 + lane;
        int row = c / 40;
        int q = c - row * 40;
        const __bf16* g;
        int byte;
        if (q < 16)      { g = xbf + (size_t)sdst[wave][row] * Cc + q * 8;       byte = q * 16; }
        else if (q < 32) { g = xbf + (size_t)ssrc[wave][row] * Cc + (q - 16) * 8; byte = q * 16; }
        else             { g = eabf + (size_t)(e0 + row) * Dd + (q - 32) * 8;    byte = 512 + (q - 32) * 16; }
        async_copy_b128(ldsbase + row * 656 + byte, g);
    }
    async_wait0();

    // ---- WMMA: 8 N-tiles (128 out channels), K = 320 = 10 chunks of 32 ----
    const int arow  = lane & 15;
    const int abase = (lane >> 4) * 8;
    const int ncol0 = lane & 15;
    const int rbase = (lane >> 4) * 8;
    const __bf16* zrow = &zbuf[wave][arow][0];

    for (int nt = 0; nt < 8; ++nt) {
        v8f gacc = {};
        v8f cacc = {};
        const __bf16* pg = packFS + ((size_t)((layer * 2 + 0) * 8 + nt) * 10) * 512 + lane * 16;
        const __bf16* ps = packFS + ((size_t)((layer * 2 + 1) * 8 + nt) * 10) * 512 + lane * 16;
#pragma unroll
        for (int kc = 0; kc < 10; ++kc) {
            ABfrag A, Bg, Bs;
            A.h[0] = *(const v8bf*)(zrow + kc * 32 + abase);
            A.h[1] = *(const v8bf*)(zrow + kc * 32 + abase + 16);
            Bg.h[0] = *(const v8bf*)(pg + (size_t)kc * 512);
            Bg.h[1] = *(const v8bf*)(pg + (size_t)kc * 512 + 8);
            Bs.h[0] = *(const v8bf*)(ps + (size_t)kc * 512);
            Bs.h[1] = *(const v8bf*)(ps + (size_t)kc * 512 + 8);
            gacc = __builtin_amdgcn_wmma_f32_16x16x32_bf16(false, A.v, false, Bg.v,
                                                           (short)0, gacc, false, false);
            cacc = __builtin_amdgcn_wmma_f32_16x16x32_bf16(false, A.v, false, Bs.v,
                                                           (short)0, cacc, false, false);
        }
        const int ncol = nt * 16 + ncol0;
        const float bg = bfv[layer * Cc + ncol];
        const float bs = bsv[layer * Cc + ncol];
#pragma unroll
        for (int rr = 0; rr < 8; ++rr) {
            float g = 1.f / (1.f + __expf(-(gacc[rr] + bg)));
            float t = cacc[rr] + bs;
            float s = (t > 20.f) ? t : log1pf(__expf(t));
            int node = sdst[wave][rbase + rr];
            atomicAdd(&agg[(size_t)node * Cc + ncol], g * s);
        }
    }
}

// ---------------------------------------------------------------------------
// GEMM1: h = x @ W1^T + b1  (M=N nodes, K=128, N=512); accumulate BN stats.
// A tile async-staged from bf16 shadow of x. One wave/tile, 4 waves/block.
// ---------------------------------------------------------------------------
__global__ void gemm1_kernel(const __bf16* __restrict__ xbf, const __bf16* __restrict__ packW1,
                             const float* __restrict__ b1, float* __restrict__ hbuf,
                             float* __restrict__ ssum, float* __restrict__ ssq, int layer) {
    const int wave = threadIdx.x >> 5;
    const int lane = threadIdx.x & 31;
    const int tile = blockIdx.x * 4 + wave;
    __shared__ __bf16 xs[4][16][136];     // 136 halves = 272 B per row (17 x 16B)
    if (tile >= Nn / 16) return;
    const int n0 = tile * 16;

    // ---- async-stage x tile: 16 rows x 16 chunks of 16B = 256 chunks ----
    const unsigned ldsbase = (unsigned)(size_t)&xs[wave][0][0];
#pragma unroll
    for (int t = 0; t < 8; ++t) {
        int c = t * 32 + lane;
        int row = c >> 4, q = c & 15;
        async_copy_b128(ldsbase + row * 272 + q * 16,
                        xbf + (size_t)(n0 + row) * Cc + q * 8);
    }
    async_wait0();

    const int arow = lane & 15, abase = (lane >> 4) * 8;
    ABfrag A[4];
#pragma unroll
    for (int kc = 0; kc < 4; ++kc) {
        A[kc].h[0] = *(const v8bf*)&xs[wave][arow][kc * 32 + abase];
        A[kc].h[1] = *(const v8bf*)&xs[wave][arow][kc * 32 + abase + 16];
    }

    const __bf16* wbase = packW1 + (size_t)layer * 32 * 4 * 512 + lane * 16;
    const int ncol0 = lane & 15, rbase = (lane >> 4) * 8;
    for (int nt = 0; nt < 32; ++nt) {
        v8f acc = {};
#pragma unroll
        for (int kc = 0; kc < 4; ++kc) {
            ABfrag B;
            const __bf16* bp = wbase + (size_t)(nt * 4 + kc) * 512;
            B.h[0] = *(const v8bf*)bp;
            B.h[1] = *(const v8bf*)(bp + 8);
            acc = __builtin_amdgcn_wmma_f32_16x16x32_bf16(false, A[kc].v, false, B.v,
                                                          (short)0, acc, false, false);
        }
        const int ncol = nt * 16 + ncol0;
        const float bias = b1[layer * Hh + ncol];
        float s = 0.f, s2 = 0.f;
#pragma unroll
        for (int rr = 0; rr < 8; ++rr) {
            float v = acc[rr] + bias;
            hbuf[(size_t)(n0 + rbase + rr) * Hh + ncol] = v;
            s += v; s2 += v * v;
        }
        atomicAdd(&ssum[ncol], s);
        atomicAdd(&ssq[ncol], s2);
    }
}

__global__ void bnfinal_kernel(const float* __restrict__ ssum, const float* __restrict__ ssq,
                               float* meanv, float* rstdv) {
    int c = blockIdx.x * 256 + threadIdx.x;
    if (c >= Hh) return;
    float m = ssum[c] / (float)Nn;
    float v = ssq[c] / (float)Nn - m * m;
    meanv[c] = m;
    rstdv[c] = rsqrtf(v + EPSf);
}

// ---------------------------------------------------------------------------
// GEMM2: x += relu(bn(h)) @ W2^T + b2  (M=N nodes, K=512, N=128)
// BN+ReLU folded into bf16 staging (needs VALU, so no async path here).
// ---------------------------------------------------------------------------
__global__ void gemm2_kernel(const float* __restrict__ hbuf, const __bf16* __restrict__ packW2,
                             const float* __restrict__ b2,
                             const float* __restrict__ g1, const float* __restrict__ be1,
                             const float* __restrict__ meanv, const float* __restrict__ rstdv,
                             float* __restrict__ xb, int layer) {
    const int wave = threadIdx.x >> 5;
    const int lane = threadIdx.x & 31;
    const int tile = blockIdx.x * 2 + wave;
    __shared__ __bf16 hs[2][16][520];
    if (tile >= Nn / 16) return;
    const int n0 = tile * 16;

    const int r = lane & 15, p = lane >> 4;
    const float* row = hbuf + (size_t)(n0 + r) * Hh + p * 256;
#pragma unroll 4
    for (int c = 0; c < 256; c += 4) {
        int col = p * 256 + c;
        v4f v  = *(const v4f*)(row + c);
        v4f mu = *(const v4f*)(meanv + col);
        v4f rs = *(const v4f*)(rstdv + col);
        v4f gg = *(const v4f*)(g1 + layer * Hh + col);
        v4f bb = *(const v4f*)(be1 + layer * Hh + col);
        v4f hn;
#pragma unroll
        for (int j = 0; j < 4; ++j)
            hn[j] = fmaxf(gg[j] * (v[j] - mu[j]) * rs[j] + bb[j], 0.f);
        store4bf(&hs[wave][r][col], hn);
    }

    const int arow = lane & 15, abase = (lane >> 4) * 8;
    v8f acc[8];
    v8f zv = {};
#pragma unroll
    for (int nt = 0; nt < 8; ++nt) acc[nt] = zv;

    const __bf16* wbase = packW2 + (size_t)layer * 8 * 16 * 512 + lane * 16;
    for (int kc = 0; kc < 16; ++kc) {
        ABfrag A;
        A.h[0] = *(const v8bf*)&hs[wave][arow][kc * 32 + abase];
        A.h[1] = *(const v8bf*)&hs[wave][arow][kc * 32 + abase + 16];
#pragma unroll
        for (int nt = 0; nt < 8; ++nt) {
            ABfrag B;
            const __bf16* bp = wbase + (size_t)(nt * 16 + kc) * 512;
            B.h[0] = *(const v8bf*)bp;
            B.h[1] = *(const v8bf*)(bp + 8);
            acc[nt] = __builtin_amdgcn_wmma_f32_16x16x32_bf16(false, A.v, false, B.v,
                                                              (short)0, acc[nt], false, false);
        }
    }
    const int ncol0 = lane & 15, rbase = (lane >> 4) * 8;
#pragma unroll
    for (int nt = 0; nt < 8; ++nt) {
        const int ncol = nt * 16 + ncol0;
        const float bias = b2[layer * Cc + ncol];
#pragma unroll
        for (int rr = 0; rr < 8; ++rr) {
            size_t idx = (size_t)(n0 + rbase + rr) * Cc + ncol;
            xb[idx] = xb[idx] + acc[nt][rr] + bias;
        }
    }
}

// ---------------------------------------------------------------------------
// Graph LayerNorm helpers (node_batch sorted; G=32)
// ---------------------------------------------------------------------------
__global__ void nodesum_kernel(const float* __restrict__ xb, const int* __restrict__ nb,
                               float* gsum) {
    int node = blockIdx.x * 8 + (threadIdx.x >> 5);
    int lane = threadIdx.x & 31;
    if (node >= Nn) return;
    v4f v = *(const v4f*)(xb + (size_t)node * Cc + lane * 4);
    float s = v[0] + v[1] + v[2] + v[3];
#pragma unroll
    for (int o = 16; o > 0; o >>= 1) s += __shfl_down(s, o, 32);
    if (lane == 0) atomicAdd(&gsum[nb[node]], s);
}
__global__ void gmean_kernel(const float* gsum, const float* cnt, float* gmean) {
    int g = threadIdx.x;
    if (g < Gg) gmean[g] = gsum[g] / (fmaxf(cnt[g], 1.f) * (float)Cc);
}
__global__ void nodevar_kernel(const float* __restrict__ xb, const int* __restrict__ nb,
                               const float* __restrict__ gmean, float* vsum) {
    int node = blockIdx.x * 8 + (threadIdx.x >> 5);
    int lane = threadIdx.x & 31;
    if (node >= Nn) return;
    float mu = gmean[nb[node]];
    v4f v = *(const v4f*)(xb + (size_t)node * Cc + lane * 4);
    float s = 0.f;
#pragma unroll
    for (int j = 0; j < 4; ++j) { float d = v[j] - mu; s += d * d; }
#pragma unroll
    for (int o = 16; o > 0; o >>= 1) s += __shfl_down(s, o, 32);
    if (lane == 0) atomicAdd(&vsum[nb[node]], s);
}
__global__ void grstd_kernel(const float* vsum, const float* cnt, float* grstd) {
    int g = threadIdx.x;
    if (g < Gg) grstd[g] = rsqrtf(vsum[g] / (fmaxf(cnt[g], 1.f) * (float)Cc) + EPSf);
}
// LayerNorm epilogue; also writes the bf16 shadow used by next layer's edge GEMM
__global__ void lnorm_kernel(const float* __restrict__ xb, const int* __restrict__ nb,
                             const float* __restrict__ gmean, const float* __restrict__ grstd,
                             const float* __restrict__ lnw, const float* __restrict__ lnb,
                             float* __restrict__ dst, __bf16* __restrict__ xbf, int layer) {
    int i = blockIdx.x * 256 + threadIdx.x;
    if (i >= Nn * Cc) return;
    int node = i >> 7;
    int c = i & 127;
    int b = nb[node];
    float v = (xb[i] - gmean[b]) * grstd[b] * lnw[layer * Cc + c] + lnb[layer * Cc + c];
    dst[i] = v;
    xbf[i] = f2bf(v);
}

// ---------------------------------------------------------------------------
extern "C" void kernel_launch(void* const* d_in, const int* in_sizes, int n_in,
                              void* d_out, int out_size, void* d_ws, size_t ws_size,
                              hipStream_t stream) {
    (void)in_sizes; (void)n_in; (void)out_size; (void)ws_size;
    const float* x    = (const float*)d_in[0];
    const int*   nb   = (const int*)d_in[1];
    const int*   eidx = (const int*)d_in[2];
    const float* ea   = (const float*)d_in[3];
    const float* Wf   = (const float*)d_in[4];
    const float* bfv  = (const float*)d_in[5];
    const float* Ws   = (const float*)d_in[6];
    const float* bsv  = (const float*)d_in[7];
    const float* W1   = (const float*)d_in[8];
    const float* b1   = (const float*)d_in[9];
    const float* g1   = (const float*)d_in[10];
    const float* be1  = (const float*)d_in[11];
    const float* W2   = (const float*)d_in[12];
    const float* b2   = (const float*)d_in[13];
    const float* lnw  = (const float*)d_in[14];
    const float* lnb  = (const float*)d_in[15];
    float* out = (float*)d_out;

    char* ws = (char*)d_ws;
    size_t off = 0;
    auto alloc = [&](size_t bytes) -> char* {
        char* p = ws + off;
        off += (bytes + 255) & ~(size_t)255;
        return p;
    };
    __bf16* packFS = (__bf16*)alloc((size_t)Ll * 2 * Cc * Zz * 2);
    __bf16* packW1 = (__bf16*)alloc((size_t)Ll * Hh * Cc * 2);
    __bf16* packW2 = (__bf16*)alloc((size_t)Ll * Cc * Hh * 2);
    float*  xbuf   = (float*)alloc((size_t)Nn * Cc * 4);
    __bf16* xbf    = (__bf16*)alloc((size_t)Nn * Cc * 2);
    float*  agg    = (float*)alloc((size_t)Nn * Cc * 4);
    float*  hbuf   = (float*)alloc((size_t)Nn * Hh * 4);
    __bf16* eabf   = (__bf16*)alloc((size_t)Ee * Dd * 2);
    float* ssum  = (float*)alloc(Hh * 4);
    float* ssq   = (float*)alloc(Hh * 4);
    float* meanv = (float*)alloc(Hh * 4);
    float* rstdv = (float*)alloc(Hh * 4);
    float* cntb  = (float*)alloc(Gg * 4);
    float* gsum  = (float*)alloc(Gg * 4);
    float* vsum  = (float*)alloc(Gg * 4);
    float* gmu   = (float*)alloc(Gg * 4);
    float* grs   = (float*)alloc(Gg * 4);

    // pack all weights into WMMA B-fragment layout (bf16)
    {
        int tCZ = Cc * Zz, tHC = Hh * Cc;
        for (int l = 0; l < Ll; ++l) {
            pack_b_kernel<<<(tCZ + 255) / 256, 256, 0, stream>>>(
                Wf + (size_t)l * tCZ, packFS + ((size_t)l * 2 + 0) * tCZ, Cc, Zz);
            pack_b_kernel<<<(tCZ + 255) / 256, 256, 0, stream>>>(
                Ws + (size_t)l * tCZ, packFS + ((size_t)l * 2 + 1) * tCZ, Cc, Zz);
            pack_b_kernel<<<(tHC + 255) / 256, 256, 0, stream>>>(
                W1 + (size_t)l * tHC, packW1 + (size_t)l * tHC, Hh, Cc);
            pack_b_kernel<<<(tHC + 255) / 256, 256, 0, stream>>>(
                W2 + (size_t)l * tHC, packW2 + (size_t)l * tHC, Cc, Hh);
        }
    }

    hipMemcpyAsync(xbuf, x, (size_t)Nn * Cc * 4, hipMemcpyDeviceToDevice, stream);
    cvt_bf16_kernel<<<(Nn * Cc / 4 + 255) / 256, 256, 0, stream>>>(x, xbf, Nn * Cc / 4);
    cvt_bf16_kernel<<<(Ee * Dd / 4 + 255) / 256, 256, 0, stream>>>(ea, eabf, Ee * Dd / 4);

    zero_kernel<<<1, 256, 0, stream>>>(cntb, Gg);
    count_kernel<<<(Nn + 255) / 256, 256, 0, stream>>>(nb, cntb);

    const int NC = Nn * Cc;
    for (int l = 0; l < Ll; ++l) {
        zero_kernel<<<(NC + 255) / 256, 256, 0, stream>>>(agg, NC);
        zero_kernel<<<2, 256, 0, stream>>>(ssum, Hh);
        zero_kernel<<<2, 256, 0, stream>>>(ssq, Hh);
        zero_kernel<<<1, 256, 0, stream>>>(gsum, Gg);
        zero_kernel<<<1, 256, 0, stream>>>(vsum, Gg);

        // CGConv edge messages + scatter-add (WMMA bf16, async-to-LDS staging)
        edge_kernel<<<(Ee / 16) / 4, 128, 0, stream>>>(xbf, eabf, eidx, packFS, bfv, bsv, agg, l);
        addagg_kernel<<<(NC + 255) / 256, 256, 0, stream>>>(xbuf, xbf, agg, NC);

        // MLP with BatchNorm between (WMMA)
        gemm1_kernel<<<(Nn / 16 + 3) / 4, 128, 0, stream>>>(xbf, packW1, b1, hbuf, ssum, ssq, l);
        bnfinal_kernel<<<2, 256, 0, stream>>>(ssum, ssq, meanv, rstdv);
        gemm2_kernel<<<(Nn / 16) / 2, 64, 0, stream>>>(hbuf, packW2, b2, g1, be1,
                                                       meanv, rstdv, xbuf, l);

        // graph-mode LayerNorm
        nodesum_kernel<<<Nn / 8, 256, 0, stream>>>(xbuf, nb, gsum);
        gmean_kernel<<<1, 32, 0, stream>>>(gsum, cntb, gmu);
        nodevar_kernel<<<Nn / 8, 256, 0, stream>>>(xbuf, nb, gmu, vsum);
        grstd_kernel<<<1, 32, 0, stream>>>(vsum, cntb, grs);
        float* dst = (l == Ll - 1) ? out : xbuf;
        lnorm_kernel<<<(NC + 255) / 256, 256, 0, stream>>>(xbuf, nb, gmu, grs, lnw, lnb,
                                                           dst, xbf, l);
    }
}
